// EnTransformerLayer_vel_25589415150120
// MI455X (gfx1250) — matrix operand order repaired and checked by reading, not compiled, not run on hardware
//
#include <hip/hip_runtime.h>
#include <math.h>
#include <stdint.h>

#define NN 50000
#define EE 640000

typedef __attribute__((ext_vector_type(16))) _Float16 v16h;
typedef __attribute__((ext_vector_type(8)))  _Float16 v8h;
typedef __attribute__((ext_vector_type(8)))  float    v8f;

__device__ __forceinline__ int laneid() { return threadIdx.x & 31; }

// A-fragment (16x32 f16, ISA 7.12.2): lane L -> row L%16,
// elems 0..7 = K k0+ph*8.., elems 8..15 = K k0+16+ph*8..   (ph = L>>4)
__device__ __forceinline__ v16h frag_a(const _Float16* base, int ld, int k0) {
  int lane = laneid();
  const _Float16* p = base + (size_t)(lane & 15) * ld + k0 + ((lane >> 4) << 3);
  v8h lo = *(const v8h*)(p);
  v8h hi = *(const v8h*)(p + 16);
  return __builtin_shufflevector(lo, hi, 0,1,2,3,4,5,6,7,8,9,10,11,12,13,14,15);
}

// B-fragment (32x16 f16): lane L -> col L%16,
// elems 0..7 = K k0+ph*8.., elems 8..15 = K k0+16+ph*8..  bt is B^T row-major: bt[col][k].
__device__ __forceinline__ v16h frag_b(const _Float16* bt, int ld, int k0) {
  int lane = laneid();
  const _Float16* p = bt + (size_t)(lane & 15) * ld + k0 + ((lane >> 4) << 3);
  v8h lo = *(const v8h*)(p);
  v8h hi = *(const v8h*)(p + 16);
  return __builtin_shufflevector(lo, hi, 0,1,2,3,4,5,6,7,8,9,10,11,12,13,14,15);
}

__device__ __forceinline__ v8f wmma32(v16h a, v16h b, v8f c) {
  return __builtin_amdgcn_wmma_f32_16x16x32_f16(false, a, false, b, (short)0, c, false, false);
}

__device__ __forceinline__ float siluf(float v) { return v / (1.0f + __expf(-v)); }

// async 16B copy global -> LDS (gfx1250 VGLOBAL async op, tracked on ASYNCcnt)
__device__ __forceinline__ void async_copy16(uint32_t lds_off, const void* gaddr) {
  asm volatile("global_load_async_to_lds_b128 %0, %1, off"
               :: "v"(lds_off), "v"(gaddr) : "memory");
}
__device__ __forceinline__ void wait_async0() {
  asm volatile("s_wait_asynccnt 0x0" ::: "memory");
}

// ---------------- weight convert: dst[n*ld + k] = (f16) src[k*Ncol + n] -------------
__global__ void wconv_kernel(const float* __restrict__ src, _Float16* __restrict__ dst,
                             int K, int Ncol, int ld) {
  int i = blockIdx.x * blockDim.x + threadIdx.x;
  if (i >= K * Ncol) return;
  int k = i / Ncol, n = i % Ncol;
  dst[(size_t)n * ld + k] = (_Float16)src[i];
}

// ---------------- LayerNorm(128) -> f16, one wave per row ---------------------------
__global__ void ln_kernel(const float* __restrict__ in, const float* __restrict__ g,
                          const float* __restrict__ b, _Float16* __restrict__ out, int n) {
  int row = (blockIdx.x * blockDim.x + threadIdx.x) >> 5;
  int lane = laneid();
  if (row >= n) return;
  const float* rp = in + (size_t)row * 128;
  float4 v = *(const float4*)(rp + lane * 4);
  float s  = v.x + v.y + v.z + v.w;
  float sq = v.x*v.x + v.y*v.y + v.z*v.z + v.w*v.w;
  for (int o = 16; o > 0; o >>= 1) { s += __shfl_xor(s, o); sq += __shfl_xor(sq, o); }
  float m = s * (1.0f/128.0f);
  float var = sq * (1.0f/128.0f) - m * m;
  float inv = rsqrtf(var + 1e-5f);
  float vv[4] = {v.x, v.y, v.z, v.w};
  for (int j = 0; j < 4; ++j) {
    int kk = lane * 4 + j;
    out[(size_t)row * 128 + kk] = (_Float16)((vv[j] - m) * inv * g[kk] + b[kk]);
  }
}

// -------- generic WMMA GEMM, 4 m-subtiles per wave (B-fragment reuse x4) ------------
// out = [silu](A @ B + bias) [+ addend]
__global__ __launch_bounds__(256)
void gemm_wmma_kernel(const _Float16* __restrict__ A, int lda,
                      const _Float16* __restrict__ Bt, int ldb,
                      const float* __restrict__ bias,
                      const float* __restrict__ addend,
                      float* __restrict__ outF32, _Float16* __restrict__ outF16,
                      int ldo, int ntiles, int ksteps, int totalTiles, int applySilu,
                      int mrows) {
  int wave = blockIdx.x * (blockDim.x >> 5) + (threadIdx.x >> 5);
  if (wave >= totalTiles) return;
  int mchunk = wave / ntiles, ntile = wave % ntiles;
  int lane = laneid();
  const _Float16* Ab = A  + (size_t)mchunk * 64 * lda;
  const _Float16* Bb = Bt + (size_t)ntile * 16 * ldb;
  v8f acc[4] = {{}, {}, {}, {}};
  for (int ks = 0; ks < ksteps; ++ks) {
    v16h b = frag_b(Bb, ldb, ks * 32);
    acc[0] = wmma32(frag_a(Ab,            lda, ks * 32), b, acc[0]);
    acc[1] = wmma32(frag_a(Ab + 16 * lda, lda, ks * 32), b, acc[1]);
    acc[2] = wmma32(frag_a(Ab + 32 * lda, lda, ks * 32), b, acc[2]);
    acc[3] = wmma32(frag_a(Ab + 48 * lda, lda, ks * 32), b, acc[3]);
  }
  int col = lane & 15, ph = lane >> 4;
  int gn = ntile * 16 + col;
  float bv = bias ? bias[gn] : 0.0f;
  for (int ms = 0; ms < 4; ++ms) {
    int mbase = mchunk * 64 + ms * 16;
    if (mbase >= mrows) break;
    for (int r = 0; r < 8; ++r) {
      int gm = mbase + r + ph * 8;
      float v = acc[ms][r] + bv;
      if (applySilu) v = siluf(v);
      if (addend) v += addend[(size_t)gm * ldo + gn];
      if (outF32) outF32[(size_t)gm * ldo + gn] = v;
      else        outF16[(size_t)gm * ldo + gn] = (_Float16)v;
    }
  }
}

// ---------------- per-node head-to-head attention, one wave per node ----------------
__global__ void attn_kernel(const _Float16* __restrict__ qkv, _Float16* __restrict__ feats,
                            int n) {
  int node = (blockIdx.x * blockDim.x + threadIdx.x) >> 5;
  int lane = laneid();
  if (node >= n) return;
  const _Float16* base = qkv + (size_t)node * 768;
  int hh = (lane >> 2) & 3, jj = lane & 3;
  float dot = 0.0f;
  {
    const _Float16* qp = base + hh * 64;
    const _Float16* kp = base + 256 + jj * 64;
    for (int t = 0; t < 64; t += 8) {
      v8h a = *(const v8h*)(qp + t);
      v8h b = *(const v8h*)(kp + t);
      for (int u = 0; u < 8; ++u) dot += (float)a[u] * (float)b[u];
    }
    dot *= 0.125f;  // 64^-0.5
  }
  float m = dot;
  m = fmaxf(m, __shfl_xor(m, 1)); m = fmaxf(m, __shfl_xor(m, 2));
  float p = __expf(dot - m);
  float s = p;
  s += __shfl_xor(s, 1); s += __shfl_xor(s, 2);
  float attn = p / s;
  int h2 = lane >> 3, d8 = lane & 7;
  float f[8] = {0,0,0,0,0,0,0,0};
  for (int j = 0; j < 4; ++j) {
    float a = __shfl(attn, h2 * 4 + j);
    v8h vv = *(const v8h*)(base + 512 + j * 64 + d8 * 8);
    for (int u = 0; u < 8; ++u) f[u] += a * (float)vv[u];
  }
  v8h ov;
  for (int u = 0; u < 8; ++u) ov[u] = (_Float16)f[u];
  *(v8h*)(feats + (size_t)node * 256 + h2 * 64 + d8 * 8) = ov;
}

// ---------------- vel_scale = silu(hn @ Wv1 + bv1) . Wv2 + bv2 ----------------------
__global__ __launch_bounds__(256)
void velscale_kernel(const _Float16* __restrict__ hn16, const _Float16* __restrict__ Wv1t,
                     const float* __restrict__ bv1, const float* __restrict__ Wv2,
                     const float* __restrict__ bv2, float* __restrict__ vscale, int mtiles) {
  int wave = blockIdx.x * (blockDim.x >> 5) + (threadIdx.x >> 5);
  if (wave >= mtiles) return;
  int lane = laneid(), col = lane & 15, ph = lane >> 4;
  const _Float16* Ab = hn16 + (size_t)wave * 16 * 128;
  float part[8] = {0,0,0,0,0,0,0,0};
  for (int nt = 0; nt < 8; ++nt) {
    v8f acc = {};
    const _Float16* Bb = Wv1t + (size_t)nt * 16 * 128;
    for (int ks = 0; ks < 4; ++ks)
      acc = wmma32(frag_a(Ab, 128, ks * 32), frag_b(Bb, 128, ks * 32), acc);
    int gn = nt * 16 + col;
    float bb = bv1[gn], w2 = Wv2[gn];
    for (int r = 0; r < 8; ++r) part[r] += siluf(acc[r] + bb) * w2;
  }
  for (int o = 1; o < 16; o <<= 1)
    for (int r = 0; r < 8; ++r) part[r] += __shfl_xor(part[r], o);
  if ((lane & 15) == 0)
    for (int r = 0; r < 8; ++r) vscale[wave * 16 + r + ph * 8] = part[r] + bv2[0];
}

// ---- edge pipeline: async gather -> MLP1 -> MLP2 -> coorsMLP -> ew -----------------
// 32 edges per wave (2 m-subtiles share each B fragment), 2 waves per block.
#define EWV 2
__global__ __launch_bounds__(64)
void edge_kernel(const _Float16* __restrict__ hn16, const float* __restrict__ x,
                 const int* __restrict__ eidx,
                 const _Float16* __restrict__ We1t, const _Float16* __restrict__ We2t,
                 const _Float16* __restrict__ Wc1t, const float* __restrict__ We1_last,
                 const float* __restrict__ be1, const float* __restrict__ be2,
                 const float* __restrict__ bc1, const float* __restrict__ Wc2,
                 const float* __restrict__ bc2, float* __restrict__ ew) {
  __shared__ __align__(16) _Float16 Abuf[EWV][32 * 264];
  __shared__ __align__(16) _Float16 Ebuf[EWV][32 * 128];
  __shared__ float Dist[EWV][32];
  int wv = threadIdx.x >> 5, lane = laneid();
  int tile = blockIdx.x * EWV + wv;          // 32 edges per tile, E divisible
  _Float16* Aw = Abuf[wv];
  _Float16* Ef = Ebuf[wv];
  int col = lane & 15, ph = lane >> 4;

  int e = tile * 32 + lane;
  int r = eidx[e], c = eidx[EE + e];
  {
    float dx = x[r*3+0] - x[c*3+0];
    float dy = x[r*3+1] - x[c*3+1];
    float dz = x[r*3+2] - x[c*3+2];
    Dist[wv][lane] = dx*dx + dy*dy + dz*dz;
  }
  // gather A tile rows: [0..127]=hn[row], [128..255]=hn[col], via async LDS DMA
  for (int e2 = 0; e2 < 32; ++e2) {
    int rr = __shfl(r, e2), cc = __shfl(c, e2);
    const _Float16* src = (lane < 16) ? (hn16 + (size_t)rr * 128 + lane * 8)
                                      : (hn16 + (size_t)cc * 128 + (lane - 16) * 8);
    uint32_t dst = (uint32_t)(uintptr_t)(Aw + e2 * 264 + lane * 8);
    async_copy16(dst, src);
  }
  wait_async0();
  __syncthreads();

  // GEMM1: (32x256) @ We1 (+rank-1 rel_dist col 256) + be1, silu -> Ef (32x128)
  for (int nt = 0; nt < 8; ++nt) {
    v8f a0 = {}, a1 = {};
    const _Float16* Bb = We1t + (size_t)nt * 16 * 264;
    for (int ks = 0; ks < 8; ++ks) {
      v16h b = frag_b(Bb, 264, ks * 32);
      a0 = wmma32(frag_a(Aw,            264, ks * 32), b, a0);
      a1 = wmma32(frag_a(Aw + 16 * 264, 264, ks * 32), b, a1);
    }
    int gn = nt * 16 + col;
    float wl = We1_last[gn], bb = be1[gn];
    for (int rr2 = 0; rr2 < 8; ++rr2) {
      int m0 = rr2 + ph * 8;
      Ef[m0 * 128 + gn]        = (_Float16)siluf(a0[rr2] + Dist[wv][m0] * wl + bb);
      Ef[(16 + m0) * 128 + gn] = (_Float16)siluf(a1[rr2] + Dist[wv][16 + m0] * wl + bb);
    }
  }
  __syncthreads();

  // GEMM2: Ef(32x128) @ We2 + be2, silu -> Abuf reused as (32x128)
  for (int nt = 0; nt < 8; ++nt) {
    v8f a0 = {}, a1 = {};
    const _Float16* Bb = We2t + (size_t)nt * 16 * 128;
    for (int ks = 0; ks < 4; ++ks) {
      v16h b = frag_b(Bb, 128, ks * 32);
      a0 = wmma32(frag_a(Ef,            128, ks * 32), b, a0);
      a1 = wmma32(frag_a(Ef + 16 * 128, 128, ks * 32), b, a1);
    }
    int gn = nt * 16 + col;
    float bb = be2[gn];
    for (int rr2 = 0; rr2 < 8; ++rr2) {
      int m0 = rr2 + ph * 8;
      Aw[m0 * 128 + gn]        = (_Float16)siluf(a0[rr2] + bb);
      Aw[(16 + m0) * 128 + gn] = (_Float16)siluf(a1[rr2] + bb);
    }
  }
  __syncthreads();

  // GEMM3: Aw(32x128) @ Wc1 + bc1, silu -> Ef
  for (int nt = 0; nt < 8; ++nt) {
    v8f a0 = {}, a1 = {};
    const _Float16* Bb = Wc1t + (size_t)nt * 16 * 128;
    for (int ks = 0; ks < 4; ++ks) {
      v16h b = frag_b(Bb, 128, ks * 32);
      a0 = wmma32(frag_a(Aw,            128, ks * 32), b, a0);
      a1 = wmma32(frag_a(Aw + 16 * 128, 128, ks * 32), b, a1);
    }
    int gn = nt * 16 + col;
    float bb = bc1[gn];
    for (int rr2 = 0; rr2 < 8; ++rr2) {
      int m0 = rr2 + ph * 8;
      Ef[m0 * 128 + gn]        = (_Float16)siluf(a0[rr2] + bb);
      Ef[(16 + m0) * 128 + gn] = (_Float16)siluf(a1[rr2] + bb);
    }
  }
  __syncthreads();

  // ew[e] = Ef[e,:] . Wc2 + bc2  (all 32 lanes, one edge each)
  {
    float s = bc2[0];
    for (int kk = 0; kk < 128; ++kk) s += (float)Ef[lane * 128 + kk] * Wc2[kk];
    ew[tile * 32 + lane] = s;
  }
}

// ---------------- global softmax reductions over ew ---------------------------------
__global__ void redmax_part(const float* __restrict__ ew, int n, float* __restrict__ part) {
  __shared__ float sm[8];
  float m = -3.4e38f;
  for (int i = blockIdx.x * blockDim.x + threadIdx.x; i < n; i += gridDim.x * blockDim.x)
    m = fmaxf(m, ew[i]);
  for (int o = 16; o > 0; o >>= 1) m = fmaxf(m, __shfl_xor(m, o));
  if ((threadIdx.x & 31) == 0) sm[threadIdx.x >> 5] = m;
  __syncthreads();
  if (threadIdx.x == 0) {
    float mm = sm[0];
    for (int i = 1; i < (int)(blockDim.x >> 5); ++i) mm = fmaxf(mm, sm[i]);
    part[blockIdx.x] = mm;
  }
}
__global__ void redmax_final(const float* __restrict__ part, int n, float* __restrict__ scal) {
  __shared__ float sm[8];
  float m = -3.4e38f;
  for (int i = threadIdx.x; i < n; i += blockDim.x) m = fmaxf(m, part[i]);
  for (int o = 16; o > 0; o >>= 1) m = fmaxf(m, __shfl_xor(m, o));
  if ((threadIdx.x & 31) == 0) sm[threadIdx.x >> 5] = m;
  __syncthreads();
  if (threadIdx.x == 0) {
    float mm = sm[0];
    for (int i = 1; i < (int)(blockDim.x >> 5); ++i) mm = fmaxf(mm, sm[i]);
    scal[0] = mm;
  }
}
__global__ void redsum_part(const float* __restrict__ ew, int n, const float* __restrict__ scal,
                            float* __restrict__ part) {
  __shared__ float sm[8];
  float mx = scal[0];
  float s = 0.0f;
  for (int i = blockIdx.x * blockDim.x + threadIdx.x; i < n; i += gridDim.x * blockDim.x)
    s += __expf(ew[i] - mx);
  for (int o = 16; o > 0; o >>= 1) s += __shfl_xor(s, o);
  if ((threadIdx.x & 31) == 0) sm[threadIdx.x >> 5] = s;
  __syncthreads();
  if (threadIdx.x == 0) {
    float ss = 0.0f;
    for (int i = 0; i < (int)(blockDim.x >> 5); ++i) ss += sm[i];
    part[blockIdx.x] = ss;
  }
}
__global__ void redsum_final(const float* __restrict__ part, int n, float* __restrict__ scal) {
  __shared__ float sm[8];
  float s = 0.0f;
  for (int i = threadIdx.x; i < n; i += blockDim.x) s += part[i];
  for (int o = 16; o > 0; o >>= 1) s += __shfl_xor(s, o);
  if ((threadIdx.x & 31) == 0) sm[threadIdx.x >> 5] = s;
  __syncthreads();
  if (threadIdx.x == 0) {
    float ss = 0.0f;
    for (int i = 0; i < (int)(blockDim.x >> 5); ++i) ss += sm[i];
    scal[1] = ss;
  }
}

__global__ void zero_kernel(float* __restrict__ p, int n) {
  int i = blockIdx.x * blockDim.x + threadIdx.x;
  if (i < n) p[i] = 0.0f;
}

__global__ void scatter_kernel(const float* __restrict__ ew, const int* __restrict__ eidx,
                               const float* __restrict__ x, const float* __restrict__ scal,
                               float* __restrict__ coor) {
  int e = blockIdx.x * blockDim.x + threadIdx.x;
  if (e >= EE) return;
  float w = __expf(ew[e] - scal[0]) / scal[1];
  int r = eidx[e], c = eidx[EE + e];
  atomicAdd(&coor[r*3+0], w * (x[r*3+0] - x[c*3+0]));
  atomicAdd(&coor[r*3+1], w * (x[r*3+1] - x[c*3+1]));
  atomicAdd(&coor[r*3+2], w * (x[r*3+2] - x[c*3+2]));
}

__global__ void finalize_x_kernel(const float* __restrict__ x, const float* __restrict__ coor,
                                  const float* __restrict__ vscale, const float* __restrict__ vel,
                                  float* __restrict__ outx, int n3) {
  int i = blockIdx.x * blockDim.x + threadIdx.x;
  if (i >= n3) return;
  outx[i] = x[i] + coor[i] + vscale[i / 3] * vel[i];
}

// =====================================================================================
extern "C" void kernel_launch(void* const* d_in, const int* in_sizes, int n_in,
                              void* d_out, int out_size, void* d_ws, size_t ws_size,
                              hipStream_t stream) {
  // setup_inputs() dict order, params expanded in insertion order:
  const float* h     = (const float*)d_in[0];
  const float* x     = (const float*)d_in[1];
  const float* vel   = (const float*)d_in[2];
  const float* ln1_g = (const float*)d_in[3];
  const float* ln1_b = (const float*)d_in[4];
  const float* ln2_g = (const float*)d_in[5];
  const float* ln2_b = (const float*)d_in[6];
  const float* We1   = (const float*)d_in[7];
  const float* be1   = (const float*)d_in[8];
  const float* We2   = (const float*)d_in[9];
  const float* be2   = (const float*)d_in[10];
  const float* Wqkv  = (const float*)d_in[11];
  const float* Wo    = (const float*)d_in[12];
  const float* bo    = (const float*)d_in[13];
  const float* Wc1   = (const float*)d_in[14];
  const float* bc1   = (const float*)d_in[15];
  const float* Wc2   = (const float*)d_in[16];
  const float* bc2   = (const float*)d_in[17];
  const float* Wv1   = (const float*)d_in[18];
  const float* bv1   = (const float*)d_in[19];
  const float* Wv2   = (const float*)d_in[20];
  const float* bv2   = (const float*)d_in[21];
  const float* Wf1   = (const float*)d_in[22];
  const float* bf1   = (const float*)d_in[23];
  const float* Wf2   = (const float*)d_in[24];
  const float* bf2   = (const float*)d_in[25];
  const int*   eidx  = (const int*)d_in[26];

  float* out_h = (float*)d_out;                      // [N,128]
  float* out_x = out_h + (size_t)NN * 128;           // [N,3]

  char* ws = (char*)d_ws;
  size_t o = 0;
  auto alloc = [&](size_t b) { size_t c = o; o = (o + b + 255) & ~(size_t)255; return c; };
  size_t oHN16  = alloc((size_t)NN * 128 * 2);
  size_t oQKV16 = alloc((size_t)NN * 768 * 2);
  size_t oFEAT  = alloc((size_t)NN * 256 * 2);
  size_t oH1    = alloc((size_t)NN * 128 * 4);
  size_t oH2    = alloc((size_t)NN * 128 * 2);
  size_t oFF1   = alloc((size_t)NN * 256 * 2);
  size_t oEW    = alloc((size_t)EE * 4);
  size_t oCOOR  = alloc((size_t)NN * 3 * 4);
  size_t oVSC   = alloc((size_t)NN * 4);
  size_t oPART  = alloc(2048 * 4);
  size_t oSCAL  = alloc(2 * 4);
  size_t oWE1T  = alloc((size_t)128 * 264 * 2);
  size_t oWE2T  = alloc((size_t)128 * 128 * 2);
  size_t oWC1T  = alloc((size_t)128 * 128 * 2);
  size_t oWV1T  = alloc((size_t)128 * 128 * 2);
  size_t oWQKVT = alloc((size_t)768 * 128 * 2);
  size_t oWOT   = alloc((size_t)128 * 256 * 2);
  size_t oWF1T  = alloc((size_t)256 * 128 * 2);
  size_t oWF2T  = alloc((size_t)128 * 256 * 2);

  _Float16* HN16  = (_Float16*)(ws + oHN16);
  _Float16* QKV16 = (_Float16*)(ws + oQKV16);
  _Float16* FEAT  = (_Float16*)(ws + oFEAT);
  float*    H1    = (float*)(ws + oH1);
  _Float16* H2    = (_Float16*)(ws + oH2);
  _Float16* FF1   = (_Float16*)(ws + oFF1);
  float*    EW    = (float*)(ws + oEW);
  float*    COOR  = (float*)(ws + oCOOR);
  float*    VSC   = (float*)(ws + oVSC);
  float*    PART  = (float*)(ws + oPART);
  float*    SCAL  = (float*)(ws + oSCAL);
  _Float16* WE1T  = (_Float16*)(ws + oWE1T);
  _Float16* WE2T  = (_Float16*)(ws + oWE2T);
  _Float16* WC1T  = (_Float16*)(ws + oWC1T);
  _Float16* WV1T  = (_Float16*)(ws + oWV1T);
  _Float16* WQKVT = (_Float16*)(ws + oWQKVT);
  _Float16* WOT   = (_Float16*)(ws + oWOT);
  _Float16* WF1T  = (_Float16*)(ws + oWF1T);
  _Float16* WF2T  = (_Float16*)(ws + oWF2T);

  auto cdiv = [](int a, int b) { return (a + b - 1) / b; };

  // --- weight conversion (fp32 -> f16, transposed) ---
  wconv_kernel<<<cdiv(257*128,256),256,0,stream>>>(We1,  WE1T, 257, 128, 264);
  wconv_kernel<<<cdiv(128*128,256),256,0,stream>>>(We2,  WE2T, 128, 128, 128);
  wconv_kernel<<<cdiv(128*128,256),256,0,stream>>>(Wc1,  WC1T, 128, 128, 128);
  wconv_kernel<<<cdiv(128*128,256),256,0,stream>>>(Wv1,  WV1T, 128, 128, 128);
  wconv_kernel<<<cdiv(128*768,256),256,0,stream>>>(Wqkv, WQKVT,128, 768, 128);
  wconv_kernel<<<cdiv(256*128,256),256,0,stream>>>(Wo,   WOT,  256, 128, 256);
  wconv_kernel<<<cdiv(128*256,256),256,0,stream>>>(Wf1,  WF1T, 128, 256, 128);
  wconv_kernel<<<cdiv(256*128,256),256,0,stream>>>(Wf2,  WF2T, 256, 128, 256);

  // --- hn = LN1(h) -> f16 ---
  ln_kernel<<<cdiv(NN*32,256),256,0,stream>>>(h, ln1_g, ln1_b, HN16, NN);

  const int MT = NN / 16;          // 3125 m-tiles of 16 rows
  const int MC = cdiv(MT, 4);      // 782 m-chunks of 64 rows

  // --- qkv = hn @ Wqkv ---
  gemm_wmma_kernel<<<cdiv(MC*48,8),256,0,stream>>>(HN16,128, WQKVT,128, nullptr,nullptr,
                                                   nullptr, QKV16, 768, 48, 4, MC*48, 0, NN);
  // --- per-node attention -> feats f16 ---
  attn_kernel<<<cdiv(NN*32,256),256,0,stream>>>(QKV16, FEAT, NN);
  // --- H1 = h + feats @ Wo + bo ---
  gemm_wmma_kernel<<<cdiv(MC*8,8),256,0,stream>>>(FEAT,256, WOT,256, bo, h,
                                                  H1, nullptr, 128, 8, 8, MC*8, 0, NN);
  // --- vel scale ---
  velscale_kernel<<<cdiv(MT*32,256),256,0,stream>>>(HN16, WV1T, bv1, Wv2, bv2, VSC, MT);

  // --- edge pipeline -> ew ---
  edge_kernel<<<EE/(32*EWV),64,0,stream>>>(HN16, x, eidx, WE1T, WE2T, WC1T,
                                           We1 + (size_t)256*128, be1, be2, bc1,
                                           Wc2, bc2, EW);
  // --- softmax over all edges ---
  redmax_part<<<1024,256,0,stream>>>(EW, EE, PART);
  redmax_final<<<1,256,0,stream>>>(PART, 1024, SCAL);
  redsum_part<<<1024,256,0,stream>>>(EW, EE, SCAL, PART + 1024);
  redsum_final<<<1,256,0,stream>>>(PART + 1024, 1024, SCAL);
  // --- coordinate scatter + finalize x ---
  zero_kernel<<<cdiv(NN*3,256),256,0,stream>>>(COOR, NN*3);
  scatter_kernel<<<cdiv(EE,256),256,0,stream>>>(EW, eidx, x, SCAL, COOR);
  finalize_x_kernel<<<cdiv(NN*3,256),256,0,stream>>>(x, COOR, VSC, vel, out_x, NN*3);

  // --- feed-forward: h_out = H1 + silu(LN2(H1)@Wf1+bf1)@Wf2 + bf2 ---
  ln_kernel<<<cdiv(NN*32,256),256,0,stream>>>(H1, ln2_g, ln2_b, H2, NN);
  gemm_wmma_kernel<<<cdiv(MC*16,8),256,0,stream>>>(H2,128, WF1T,128, bf1, nullptr,
                                                   nullptr, FF1, 256, 16, 4, MC*16, 1, NN);
  gemm_wmma_kernel<<<cdiv(MC*8,8),256,0,stream>>>(FF1,256, WF2T,256, bf2, H1,
                                                  out_h, nullptr, 128, 8, 8, MC*8, 0, NN);

  (void)in_sizes; (void)n_in; (void)out_size; (void)ws_size;
}
